// GAT_63608465654039
// MI455X (gfx1250) — compile-verified, matrix operand order checked by reading, not run on hardware
//
#include <hip/hip_runtime.h>

typedef __attribute__((ext_vector_type(16))) __bf16 v16bf;
typedef __attribute__((ext_vector_type(8)))  float  v8f;

#define NEG_SLOPE 0.2f

__device__ __forceinline__ float lrelu(float x) { return x > 0.f ? x : NEG_SLOPE * x; }

// round-to-nearest-even f32 -> bf16
__device__ __forceinline__ unsigned short f2bf(float f) {
  unsigned u = __float_as_uint(f);
  unsigned r = u + 0x7FFFu + ((u >> 16) & 1u);
  if ((u & 0x7F800000u) == 0x7F800000u) r = u;  // inf/nan: truncate
  return (unsigned short)(r >> 16);
}

// order-preserving float <-> uint key (for atomicMax-based segment max).
// Key 0 corresponds to the most-negative encodable value, so zero-init == -inf.
__device__ __forceinline__ unsigned fkey(float f) {
  unsigned u = __float_as_uint(f);
  return (u & 0x80000000u) ? ~u : (u | 0x80000000u);
}
__device__ __forceinline__ float funkey(unsigned k) {
  unsigned u = (k & 0x80000000u) ? (k & 0x7FFFFFFFu) : ~k;
  return __uint_as_float(u);
}

// ---------------- conversion kernels ----------------

__global__ void k_cvt_bf16(const float* __restrict__ in, unsigned short* __restrict__ out,
                           int n, int relu) {
  int i = blockIdx.x * blockDim.x + threadIdx.x;
  if (i >= n) return;
  float v = in[i];
  if (relu) v = fmaxf(v, 0.f);
  out[i] = f2bf(v);
}

// W [K,F] row-major f32 -> Wt [F,K] bf16 (so B fragments load contiguously per column)
__global__ void k_wt_cvt(const float* __restrict__ W, unsigned short* __restrict__ Wt,
                         int K, int F) {
  int i = blockIdx.x * blockDim.x + threadIdx.x;
  if (i >= K * F) return;
  int k = i / F, c = i % F;
  Wt[(size_t)c * K + k] = f2bf(W[i]);
}

__global__ void k_fill_u32(unsigned* __restrict__ p, unsigned v, long n) {
  long i = (long)blockIdx.x * blockDim.x + threadIdx.x;
  if (i < n) p[i] = v;
}

// ---------------- WMMA GEMM: C[nrows,ncols] = A[nrows,128] @ Bt[ncols,128]^T + bias -------------
// One wave per 16-row strip: the four K-step A fragments are loaded once into
// registers and reused across every 16-column output tile (A L2 traffic / colTiles).

union FragU { v16bf bf; uint4 u2[2]; };

__global__ __launch_bounds__(256)
void k_gemm_bf16(const unsigned short* __restrict__ A,
                 const unsigned short* __restrict__ Bt,
                 const float* __restrict__ bias,
                 float* __restrict__ C,
                 int nrows, int ncols) {
  const int K = 128;
  int lane = threadIdx.x & 31;
  int wave = (blockIdx.x * blockDim.x + threadIdx.x) >> 5;
  int rowTiles = nrows >> 4;
  if (wave >= rowTiles) return;
  int colTiles = ncols >> 4;

  // Fragment layout (16-bit A 16x32, ISA 7.12.2):
  //   lane<16  -> row M=lane,    K = {k0..k0+7, k0+16..k0+23}
  //   lane>=16 -> row M=lane-16, K = {k0+8..k0+15, k0+24..k0+31}
  // B (from pre-transposed weights) mirrors this per output column.
  int khalf = (lane < 16) ? 0 : 8;
  int r = (wave << 4) + (lane & 15);
  const unsigned short* arow = A + (size_t)r * K + khalf;

  FragU fa[4];
#pragma unroll
  for (int ks = 0; ks < 4; ks++) {
    fa[ks].u2[0] = *(const uint4*)(arow + ks * 32);
    fa[ks].u2[1] = *(const uint4*)(arow + ks * 32 + 16);
  }

  int mbase = (wave << 4) + ((lane < 16) ? 0 : 8);
  float* crow = C + (size_t)mbase * ncols + (lane & 15);

  for (int ct = 0; ct < colTiles; ct++) {
    const unsigned short* bcol = Bt + (size_t)((ct << 4) + (lane & 15)) * K + khalf;
    v8f acc = {};
#pragma unroll
    for (int ks = 0; ks < 4; ks++) {
      FragU fb;
      fb.u2[0] = *(const uint4*)(bcol + ks * 32);
      fb.u2[1] = *(const uint4*)(bcol + ks * 32 + 16);
      acc = __builtin_amdgcn_wmma_f32_16x16x32_bf16(false, fa[ks].bf, false, fb.bf,
                                                    (short)0, acc, false, false);
    }
    float bv = bias[(ct << 4) + (lane & 15)];
#pragma unroll
    for (int i = 0; i < 8; i++)
      crow[(size_t)i * ncols + (ct << 4)] = acc[i] + bv;
  }
}

// ---------------- layer-1 edge kernels (H=4, D=32, 128 feats/row) ----------------

__global__ __launch_bounds__(256)
void k_edge_logit4(const int* __restrict__ src, const int* __restrict__ dst,
                   const float4* __restrict__ fs, const float4* __restrict__ fd,
                   const float* __restrict__ attn,      // [4,32] flat
                   float* __restrict__ logits,          // [E,4]
                   unsigned* __restrict__ mkey,         // [N,4]
                   int E) {
  int lane = threadIdx.x & 31;
  int e = (blockIdx.x * blockDim.x + threadIdx.x) >> 5;
  if (e >= E) return;
  int s = src[e], d = dst[e];
  float4 a = fs[(size_t)s * 32 + lane];
  float4 b = fd[(size_t)d * 32 + lane];
  float4 w = ((const float4*)attn)[lane];
  float p = lrelu(a.x + b.x) * w.x + lrelu(a.y + b.y) * w.y +
            lrelu(a.z + b.z) * w.z + lrelu(a.w + b.w) * w.w;
  p += __shfl_xor(p, 1, 32);
  p += __shfl_xor(p, 2, 32);
  p += __shfl_xor(p, 4, 32);   // reduce within each 8-lane head group
  int h = lane >> 3;
  if ((lane & 7) == 0) {
    logits[(size_t)e * 4 + h] = p;
    atomicMax(&mkey[(size_t)d * 4 + h], fkey(p));
  }
}

__global__ __launch_bounds__(256)
void k_edge_exp4(const int* __restrict__ dst, float* __restrict__ lg /*in:logits out:a*/,
                 const unsigned* __restrict__ mkey, float* __restrict__ ssum, int E) {
  int e = blockIdx.x * blockDim.x + threadIdx.x;
  if (e >= E) return;
  int d = dst[e];
  float4 l = ((float4*)lg)[e];
  float4 a;
  a.x = __expf(l.x - funkey(mkey[(size_t)d * 4 + 0]));
  a.y = __expf(l.y - funkey(mkey[(size_t)d * 4 + 1]));
  a.z = __expf(l.z - funkey(mkey[(size_t)d * 4 + 2]));
  a.w = __expf(l.w - funkey(mkey[(size_t)d * 4 + 3]));
  ((float4*)lg)[e] = a;
  atomicAdd(&ssum[(size_t)d * 4 + 0], a.x);
  atomicAdd(&ssum[(size_t)d * 4 + 1], a.y);
  atomicAdd(&ssum[(size_t)d * 4 + 2], a.z);
  atomicAdd(&ssum[(size_t)d * 4 + 3], a.w);
}

__global__ __launch_bounds__(256)
void k_edge_scatter4(const int* __restrict__ src, const int* __restrict__ dst,
                     const float4* __restrict__ fs, const float* __restrict__ a,
                     const float* __restrict__ ssum, float* __restrict__ out, int E) {
  int lane = threadIdx.x & 31;
  int e = (blockIdx.x * blockDim.x + threadIdx.x) >> 5;
  if (e >= E) return;
  int s = src[e], d = dst[e];
  int h = lane >> 3;
  float sv = ssum[(size_t)d * 4 + h];
  sv = (sv == 0.f) ? 1.f : sv;
  float alpha = a[(size_t)e * 4 + h] / sv;
  float4 v = fs[(size_t)s * 32 + lane];
  float* o = out + (size_t)d * 128 + lane * 4;
  atomicAdd(o + 0, alpha * v.x);
  atomicAdd(o + 1, alpha * v.y);
  atomicAdd(o + 2, alpha * v.z);
  atomicAdd(o + 3, alpha * v.w);
}

// ---------------- layer-2 edge kernels (H=1, D=64) ----------------

__global__ __launch_bounds__(256)
void k_edge_logit1(const int* __restrict__ src, const int* __restrict__ dst,
                   const float2* __restrict__ fs, const float2* __restrict__ fd,
                   const float* __restrict__ attn,  // [64]
                   float* __restrict__ logits, unsigned* __restrict__ mkey, int E) {
  int lane = threadIdx.x & 31;
  int e = (blockIdx.x * blockDim.x + threadIdx.x) >> 5;
  if (e >= E) return;
  int s = src[e], d = dst[e];
  float2 a = fs[(size_t)s * 32 + lane];
  float2 b = fd[(size_t)d * 32 + lane];
  float2 w = ((const float2*)attn)[lane];
  float p = lrelu(a.x + b.x) * w.x + lrelu(a.y + b.y) * w.y;
  p += __shfl_xor(p, 16, 32);
  p += __shfl_xor(p, 8, 32);
  p += __shfl_xor(p, 4, 32);
  p += __shfl_xor(p, 2, 32);
  p += __shfl_xor(p, 1, 32);
  if (lane == 0) {
    logits[e] = p;
    atomicMax(&mkey[d], fkey(p));
  }
}

__global__ __launch_bounds__(256)
void k_edge_exp1(const int* __restrict__ dst, float* __restrict__ lg,
                 const unsigned* __restrict__ mkey, float* __restrict__ ssum, int E) {
  int e = blockIdx.x * blockDim.x + threadIdx.x;
  if (e >= E) return;
  int d = dst[e];
  float a = __expf(lg[e] - funkey(mkey[d]));
  lg[e] = a;
  atomicAdd(&ssum[d], a);
}

__global__ __launch_bounds__(256)
void k_edge_scatter1(const int* __restrict__ src, const int* __restrict__ dst,
                     const float2* __restrict__ fs, const float* __restrict__ a,
                     const float* __restrict__ ssum, float* __restrict__ out, int E) {
  int lane = threadIdx.x & 31;
  int e = (blockIdx.x * blockDim.x + threadIdx.x) >> 5;
  if (e >= E) return;
  int s = src[e], d = dst[e];
  float sv = ssum[d];
  sv = (sv == 0.f) ? 1.f : sv;
  float alpha = a[e] / sv;
  float2 v = fs[(size_t)s * 32 + lane];
  float* o = out + (size_t)d * 64 + lane * 2;
  atomicAdd(o + 0, alpha * v.x);
  atomicAdd(o + 1, alpha * v.y);
}

// ---------------- host launcher ----------------

extern "C" void kernel_launch(void* const* d_in, const int* in_sizes, int n_in,
                              void* d_out, int out_size, void* d_ws, size_t ws_size,
                              hipStream_t stream) {
  const float* x    = (const float*)d_in[0];
  const int*   src  = (const int*)d_in[1];
  const int*   dst  = (const int*)d_in[2];
  const float* Wl1  = (const float*)d_in[3];
  const float* bl1  = (const float*)d_in[4];
  const float* Wr1  = (const float*)d_in[5];
  const float* br1  = (const float*)d_in[6];
  const float* at1  = (const float*)d_in[7];
  const float* Wl2  = (const float*)d_in[8];
  const float* bl2  = (const float*)d_in[9];
  const float* Wr2  = (const float*)d_in[10];
  const float* br2  = (const float*)d_in[11];
  const float* at2  = (const float*)d_in[12];

  const int F = 128, D2 = 64;
  const int N = in_sizes[0] / F;
  const int E = in_sizes[1];
  (void)n_in; (void)ws_size; (void)out_size;

  char* ws = (char*)d_ws;
  size_t off = 0;
  auto take = [&](size_t bytes) -> char* {
    char* p = ws + off;
    off += (bytes + 255) & ~(size_t)255;
    return p;
  };
  unsigned short* xbf  = (unsigned short*)take((size_t)N * F * 2);   // bf16 node feats (reused layer2)
  unsigned short* wlt1 = (unsigned short*)take((size_t)F * F * 2);
  unsigned short* wrt1 = (unsigned short*)take((size_t)F * F * 2);
  unsigned short* wlt2 = (unsigned short*)take((size_t)D2 * F * 2);
  unsigned short* wrt2 = (unsigned short*)take((size_t)D2 * F * 2);
  float*    fs   = (float*)take((size_t)N * F * 4);                  // layer2 uses first N*64
  float*    fd   = (float*)take((size_t)N * F * 4);
  // out1 / mkey / ssum are contiguous and all zero-initialized by one fill:
  float*    out1 = (float*)take((size_t)N * F * 4);
  unsigned* mk   = (unsigned*)take((size_t)N * 4 * 4);
  float*    sm   = (float*)take((size_t)N * 4 * 4);
  float*    lg   = (float*)take((size_t)E * 4 * 4);                  // logits -> exp(a), layer2 uses first E

  auto cdiv = [](long a, long b) { return (int)((a + b - 1) / b); };
  const int B = 256;

  // --- convert inputs to bf16 (weights pre-transposed) ---
  k_cvt_bf16<<<cdiv((long)N * F, B), B, 0, stream>>>(x, xbf, N * F, 0);
  k_wt_cvt<<<cdiv((long)F * F, B), B, 0, stream>>>(Wl1, wlt1, F, F);
  k_wt_cvt<<<cdiv((long)F * F, B), B, 0, stream>>>(Wr1, wrt1, F, F);
  k_wt_cvt<<<cdiv((long)F * D2, B), B, 0, stream>>>(Wl2, wlt2, F, D2);
  k_wt_cvt<<<cdiv((long)F * D2, B), B, 0, stream>>>(Wr2, wrt2, F, D2);

  // --- layer 1 GEMMs: fs = x@Wl1+bl1, fd = x@Wr1+br1 (one wave per 16-row strip) ---
  {
    int grid = cdiv((long)(N / 16) * 32, B);
    k_gemm_bf16<<<grid, B, 0, stream>>>(xbf, wlt1, bl1, fs, N, F);
    k_gemm_bf16<<<grid, B, 0, stream>>>(xbf, wrt1, br1, fd, N, F);
  }

  // --- layer 1 edge softmax + aggregate ---
  // single clear: out1 (N*F) + mkey (N*4, 0 == -inf key) + ssum (N*4) are contiguous
  k_fill_u32<<<cdiv((long)N * (F + 8), B), B, 0, stream>>>((unsigned*)out1, 0u, (long)N * (F + 8));

  k_edge_logit4<<<cdiv((long)E * 32, B), B, 0, stream>>>(src, dst, (const float4*)fs,
                                                         (const float4*)fd, at1, lg, mk, E);
  k_edge_exp4<<<cdiv((long)E, B), B, 0, stream>>>(dst, lg, mk, sm, E);
  k_edge_scatter4<<<cdiv((long)E * 32, B), B, 0, stream>>>(src, dst, (const float4*)fs,
                                                           lg, sm, out1, E);

  // --- relu + bf16 convert for layer 2 input ---
  k_cvt_bf16<<<cdiv((long)N * F, B), B, 0, stream>>>(out1, xbf, N * F, 1);

  // --- layer 2 GEMMs ---
  {
    int grid = cdiv((long)(N / 16) * 32, B);
    k_gemm_bf16<<<grid, B, 0, stream>>>(xbf, wlt2, bl2, fs, N, D2);
    k_gemm_bf16<<<grid, B, 0, stream>>>(xbf, wrt2, br2, fd, N, D2);
  }

  // --- layer 2 edge softmax + aggregate into d_out ---
  float* out = (float*)d_out;
  k_fill_u32<<<cdiv((long)N * 2, B), B, 0, stream>>>(mk, 0u, (long)N * 2); // mk[N] + sm reuse below
  k_fill_u32<<<cdiv((long)N, B), B, 0, stream>>>((unsigned*)sm, 0u, (long)N);
  k_fill_u32<<<cdiv((long)N * D2, B), B, 0, stream>>>((unsigned*)out, 0u, (long)N * D2);

  k_edge_logit1<<<cdiv((long)E * 32, B), B, 0, stream>>>(src, dst, (const float2*)fs,
                                                         (const float2*)fd, at2, lg, mk, E);
  k_edge_exp1<<<cdiv((long)E, B), B, 0, stream>>>(dst, lg, mk, sm, E);
  k_edge_scatter1<<<cdiv((long)E * 32, B), B, 0, stream>>>(src, dst, (const float2*)fs,
                                                           lg, sm, out, E);
}
